// MPNN_ENN_K_Set2Set_5076651344424
// MI455X (gfx1250) — matrix operand order repaired and checked by reading, not compiled, not run on hardware
//
#include <hip/hip_runtime.h>
#include <hip/hip_bf16.h>
#include <math.h>

#define Nn 10000
#define Ee 80000
#define Bb 64
#define NF 32
#define EF 16
#define Hh 64
#define TGTn 16
#define Tsteps 3
#define S2Sn 12

typedef __attribute__((ext_vector_type(16))) __bf16 v16bf;
typedef __attribute__((ext_vector_type(8)))  float  v8f;

struct B32x2 { uint4 lo, hi; };
__device__ __forceinline__ v16bf mk16(uint4 lo, uint4 hi) {
  B32x2 t{lo, hi};
  return __builtin_bit_cast(v16bf, t);
}
__device__ __forceinline__ unsigned short f2bf(float f) {
  unsigned u = __float_as_uint(f);
  unsigned r = u + 0x7fffu + ((u >> 16) & 1u);   // round-to-nearest-even
  return (unsigned short)(r >> 16);
}
__device__ __forceinline__ float sigm(float x) { return 1.0f / (1.0f + expf(-x)); }

// ---------------- embeddings ----------------
__global__ void __launch_bounds__(64) k_node_embed(
    const float* __restrict__ nf, const float* __restrict__ W,
    const float* __restrict__ bias, float* __restrict__ h)
{
  __shared__ float xs[NF];
  const int nid = blockIdx.x, t = threadIdx.x;
  if (t < NF) xs[t] = nf[(size_t)nid * NF + t];
  __syncthreads();
  float acc = bias[t];
  const float* wr = W + (size_t)t * NF;
  for (int j = 0; j < NF; ++j) acc += xs[j] * wr[j];
  h[(size_t)nid * Hh + t] = acc;
}

__global__ void __launch_bounds__(64) k_edge_embed(
    const float* __restrict__ ef, const float* __restrict__ W,
    const float* __restrict__ bias, unsigned short* __restrict__ R)
{
  __shared__ float xs[EF];
  const int e = blockIdx.x, t = threadIdx.x;
  if (t < EF) xs[t] = ef[(size_t)e * EF + t];
  __syncthreads();
  float acc = bias[t];
  const float* wr = W + (size_t)t * EF;
  for (int j = 0; j < EF; ++j) acc += xs[j] * wr[j];
  acc = fmaxf(acc, 0.0f);
  R[(size_t)e * Hh + t] = f2bf(acc);
}

__global__ void k_cvt(const float* __restrict__ src, unsigned short* __restrict__ dst, int n)
{
  int i = blockIdx.x * blockDim.x + threadIdx.x;
  if (i < n) dst[i] = f2bf(src[i]);
}

__global__ void k_zero(float* __restrict__ p, int n)
{
  int i = blockIdx.x * blockDim.x + threadIdx.x;
  if (i < n) p[i] = 0.0f;
}

__global__ void k_seg(const int* __restrict__ batch, int* __restrict__ seg)
{
  const int b = threadIdx.x;
  if (b > Bb) return;
  int lo = 0, hi = Nn;
  while (lo < hi) {
    int mid = (lo + hi) >> 1;
    if (batch[mid] < b) lo = mid + 1; else hi = mid;
  }
  seg[b] = lo;
}

// per-node bias term of the message: hbias[u][i] = sum_j b_e2[i*64+j] * h[u][j]
__global__ void __launch_bounds__(64) k_hbias(
    const float* __restrict__ h, const float* __restrict__ be2,
    float* __restrict__ hb)
{
  __shared__ float xs[Hh];
  const int u = blockIdx.x, t = threadIdx.x;
  xs[t] = h[(size_t)u * Hh + t];
  __syncthreads();
  float acc = 0.0f;
  const float* br = be2 + (size_t)t * Hh;
  for (int j = 0; j < Hh; ++j) acc += br[j] * xs[j];
  hb[(size_t)u * Hh + t] = acc;
}

// ---- fused edge-network + message + aggregate (WMMA bf16, A never materialized) ----
// D = (W2 tile)[16p x 64k] x (R^T tile)[64k x 16edges]; each lane owns one edge,
// so the j-contraction against hs[m][:] is lane-local; one xor-16 shuffle per channel.
__global__ void __launch_bounds__(256) k_msg(
    const unsigned short* __restrict__ Rbf,   // [E][64] bf16 relu(edge MLP1)
    const unsigned short* __restrict__ W2b,   // [4096][64] bf16 W_e2
    const float* __restrict__ hb,             // [N][64] precomputed bias term
    const float* __restrict__ h,              // [N][64]
    const int* __restrict__ Esrc,
    const int* __restrict__ Etgt,
    float* __restrict__ agg)                  // [N][64]
{
  __shared__ float hs[16][72];    // stride 72: 16B-aligned rows, breaks 16-way bank conflict
  __shared__ float hsb[16][72];
  __shared__ int src_s[16], tgt_s[16];
  const int e0 = blockIdx.x * 16;
  const int t  = threadIdx.x;

  if (t < 16) { src_s[t] = Esrc[e0 + t]; tgt_s[t] = Etgt[e0 + t]; }
  __syncthreads();
  for (int idx = t; idx < 16 * 64; idx += 256) {
    int m = idx >> 6, j = idx & 63;
    size_t srow = (size_t)src_s[m] * Hh;
    hs[m][j]  = h[srow + j];
    hsb[m][j] = hb[srow + j];
  }
  __syncthreads();

  const int wave = t >> 5, lane = t & 31;
  const int m = lane & 15;        // edge within tile (B/D column index)
  const int g = lane >> 4;        // lane-half

  // B operand (loop-invariant): R^T, 64x16. Lane-half g holds K block [g*16 .. g*16+15].
  const uint4* Rp = (const uint4*)(Rbf + (size_t)(e0 + m) * Hh);
  const v16bf b0 = mk16(Rp[2 * g],     Rp[2 * g + 1]);   // K 0..31
  const v16bf b1 = mk16(Rp[4 + 2 * g], Rp[5 + 2 * g]);   // K 32..63

  for (int ii = 0; ii < 8; ++ii) {
    const int i = wave * 8 + ii;                 // output channel 0..63
    float vp0 = 0.0f, vp1 = 0.0f, vp2 = 0.0f, vp3 = 0.0f;  // 4 independent chains
#pragma unroll
    for (int q = 0; q < 4; ++q) {
      const int ct = i * 4 + q;                  // column tile: cols [ct*16, ct*16+16)
      const int p  = ct * 16 + m;                // A-matrix row = W2 row (output column p)
      // A operand: 16x32 layout: lane-half g holds K block [g*16 .. g*16+15]
      const uint4* Ap = (const uint4*)(W2b + (size_t)p * Hh);
      const v16bf a0 = mk16(Ap[2 * g],     Ap[2 * g + 1]);
      const v16bf a1 = mk16(Ap[4 + 2 * g], Ap[5 + 2 * g]);
      v8f acc = {};
      acc = __builtin_amdgcn_wmma_f32_16x16x32_bf16(false, a0, false, b0, (short)0, acc, false, false);
      acc = __builtin_amdgcn_wmma_f32_16x16x32_bf16(false, a1, false, b1, (short)0, acc, false, false);
      // D element r -> p' = r + 8g, edge m; j = 16q + 8g + r
      const float* hrow = &hs[m][q * 16 + 8 * g];
      vp0 += acc[0] * hrow[0];
      vp1 += acc[1] * hrow[1];
      vp2 += acc[2] * hrow[2];
      vp3 += acc[3] * hrow[3];
      vp0 += acc[4] * hrow[4];
      vp1 += acc[5] * hrow[5];
      vp2 += acc[6] * hrow[6];
      vp3 += acc[7] * hrow[7];
    }
    float v = (vp0 + vp1) + (vp2 + vp3);
    v += __shfl_xor(v, 16, 32);                  // combine the two j half-sets
    if (g == 0) {
      atomicAdd(&agg[(size_t)tgt_s[m] * Hh + i], v + hsb[m][i]);
    }
  }
}

// ---------------- GRU update ----------------
__global__ void __launch_bounds__(256) k_gru(
    const float* __restrict__ agg, const float* __restrict__ hin,
    const float* __restrict__ Wih, const float* __restrict__ Whh,
    const float* __restrict__ bih, const float* __restrict__ bhh,
    float* __restrict__ hout)
{
  __shared__ float xs[4][64], hsS[4][64];
  const int t = threadIdx.x;
  const int nl = t >> 6, k = t & 63;
  const int node = blockIdx.x * 4 + nl;
  xs[nl][k]  = agg[(size_t)node * Hh + k];
  hsS[nl][k] = hin[(size_t)node * Hh + k];
  __syncthreads();
  float ir = bih[k], iz = bih[64 + k], inn = bih[128 + k];
  float hr = bhh[k], hz = bhh[64 + k], hn = bhh[128 + k];
  const float* wr = Wih + (size_t)k * 64;
  const float* wz = Wih + (size_t)(64 + k) * 64;
  const float* wn = Wih + (size_t)(128 + k) * 64;
  const float* ur = Whh + (size_t)k * 64;
  const float* uz = Whh + (size_t)(64 + k) * 64;
  const float* un = Whh + (size_t)(128 + k) * 64;
  for (int j = 0; j < 64; ++j) {
    float x = xs[nl][j], hv = hsS[nl][j];
    ir += x * wr[j];  iz += x * wz[j];  inn += x * wn[j];
    hr += hv * ur[j]; hz += hv * uz[j]; hn  += hv * un[j];
  }
  float r = sigm(ir + hr), z = sigm(iz + hz);
  float nno = tanhf(inn + r * hn);
  hout[(size_t)node * Hh + k] = (1.0f - z) * nno + z * hsS[nl][k];
}

// ---------------- Set2Set: LSTM cell ----------------
__global__ void __launch_bounds__(256) k_lstm(
    float* __restrict__ q_star, float* __restrict__ hh, float* __restrict__ cc,
    const float* __restrict__ Wih, const float* __restrict__ Whh,
    const float* __restrict__ bih, const float* __restrict__ bhh)
{
  __shared__ float qs[128], hsS[64], gs[256];
  const int b = blockIdx.x, t = threadIdx.x;
  if (t < 128) qs[t]  = q_star[b * 128 + t];
  if (t < 64)  hsS[t] = hh[b * 64 + t];
  __syncthreads();
  float g = bih[t] + bhh[t];
  const float* wi = Wih + (size_t)t * 128;
  const float* wh = Whh + (size_t)t * 64;
  for (int j = 0; j < 128; ++j) g += qs[j] * wi[j];
  for (int j = 0; j < 64;  ++j) g += hsS[j] * wh[j];
  gs[t] = g;
  __syncthreads();
  if (t < 64) {
    float iv = gs[t], fv = gs[64 + t], gg = gs[128 + t], ov = gs[192 + t];
    float c  = sigm(fv) * cc[b * 64 + t] + sigm(iv) * tanhf(gg);
    float hn = sigm(ov) * tanhf(c);
    cc[b * 64 + t] = c;
    hh[b * 64 + t] = hn;
    q_star[b * 128 + t] = hn;   // q_star[:, :H] = hh
  }
}

// ---------------- Set2Set: segment softmax attention ----------------
__global__ void __launch_bounds__(256) k_attn(
    const float* __restrict__ h, const float* __restrict__ hh,
    const int* __restrict__ seg, float* __restrict__ e_ws,
    float* __restrict__ q_star)
{
  __shared__ float hsS[64], red[256];
  const int b = blockIdx.x, t = threadIdx.x;
  const int s0 = seg[b], s1 = seg[b + 1];
  if (t < 64) hsS[t] = hh[b * 64 + t];
  __syncthreads();
  if (s0 >= s1) {
    if (t < 64) q_star[b * 128 + 64 + t] = 0.0f;
    return;
  }
  // pass 1: scores + max (deterministic tree reduce)
  float lmax = -1e30f;
  for (int nid = s0 + t; nid < s1; nid += 256) {
    const float* hr = h + (size_t)nid * Hh;
    float e = 0.0f;
    for (int j = 0; j < 64; ++j) e += hr[j] * hsS[j];
    e_ws[nid] = e;
    lmax = fmaxf(lmax, e);
  }
  red[t] = lmax; __syncthreads();
  for (int s = 128; s > 0; s >>= 1) { if (t < s) red[t] = fmaxf(red[t], red[t + s]); __syncthreads(); }
  const float emax = red[0];
  __syncthreads();
  // pass 2: exp + sum
  float lsum = 0.0f;
  for (int nid = s0 + t; nid < s1; nid += 256) {
    float ex = expf(e_ws[nid] - emax);
    e_ws[nid] = ex;
    lsum += ex;
  }
  red[t] = lsum; __syncthreads();
  for (int s = 128; s > 0; s >>= 1) { if (t < s) red[t] += red[t + s]; __syncthreads(); }
  const float ssum = red[0];
  __syncthreads();
  // pass 3: weighted sum -> q_star[:, H:2H]
  if (t < 64) {
    float acc = 0.0f;
    for (int nid = s0; nid < s1; ++nid)
      acc += e_ws[nid] * h[(size_t)nid * Hh + t];
    q_star[b * 128 + 64 + t] = acc / ssum;
  }
}

// ---------------- output projection ----------------
__global__ void k_out(const float* __restrict__ hh, const float* __restrict__ W,
                      const float* __restrict__ bias, float* __restrict__ out)
{
  int t = blockIdx.x * blockDim.x + threadIdx.x;
  if (t >= Bb * TGTn) return;
  int b = t >> 4, k = t & 15;
  float acc = bias[k];
  const float* wr = W + (size_t)k * Hh;
  const float* hr = hh + (size_t)b * Hh;
  for (int j = 0; j < Hh; ++j) acc += hr[j] * wr[j];
  out[t] = acc;
}

extern "C" void kernel_launch(void* const* d_in, const int* in_sizes, int n_in,
                              void* d_out, int out_size, void* d_ws, size_t ws_size,
                              hipStream_t stream)
{
  const float* nodef  = (const float*)d_in[0];
  const float* edgef  = (const float*)d_in[1];
  const int*   Esrc   = (const int*)d_in[2];
  const int*   Etgt   = (const int*)d_in[3];
  const int*   batch  = (const int*)d_in[4];
  const float* W_in   = (const float*)d_in[5];
  const float* b_in   = (const float*)d_in[6];
  const float* W_e1   = (const float*)d_in[7];
  const float* b_e1   = (const float*)d_in[8];
  const float* W_e2   = (const float*)d_in[9];
  const float* b_e2   = (const float*)d_in[10];
  const float* W_ih_g = (const float*)d_in[11];
  const float* W_hh_g = (const float*)d_in[12];
  const float* b_ih_g = (const float*)d_in[13];
  const float* b_hh_g = (const float*)d_in[14];
  const float* W_ih_l = (const float*)d_in[15];
  const float* W_hh_l = (const float*)d_in[16];
  const float* b_ih_l = (const float*)d_in[17];
  const float* b_hh_l = (const float*)d_in[18];
  const float* W_out  = (const float*)d_in[19];
  const float* b_outp = (const float*)d_in[20];
  (void)in_sizes; (void)n_in; (void)out_size; (void)ws_size;

  char* w = (char*)d_ws;
  size_t off = 0;
  auto wsalloc = [&](size_t bytes) -> void* {
    void* p = w + off;
    off = (off + bytes + 255) & ~(size_t)255;
    return p;
  };
  float* hA            = (float*)wsalloc((size_t)Nn * Hh * sizeof(float));
  float* hB            = (float*)wsalloc((size_t)Nn * Hh * sizeof(float));
  float* agg           = (float*)wsalloc((size_t)Nn * Hh * sizeof(float));
  float* hbias         = (float*)wsalloc((size_t)Nn * Hh * sizeof(float));
  unsigned short* Rbf  = (unsigned short*)wsalloc((size_t)Ee * Hh * sizeof(unsigned short));
  unsigned short* W2b  = (unsigned short*)wsalloc((size_t)Hh * Hh * Hh * sizeof(unsigned short));
  float* e_ws          = (float*)wsalloc((size_t)Nn * sizeof(float));
  float* hh            = (float*)wsalloc((size_t)Bb * Hh * sizeof(float));
  float* ccv           = (float*)wsalloc((size_t)Bb * Hh * sizeof(float));
  float* q_star        = (float*)wsalloc((size_t)Bb * 2 * Hh * sizeof(float));
  int*   seg           = (int*)wsalloc((size_t)(Bb + 1) * sizeof(int));

  // embeddings + bf16 weight conversion + batch segment starts
  k_node_embed<<<Nn, 64, 0, stream>>>(nodef, W_in, b_in, hA);
  k_edge_embed<<<Ee, 64, 0, stream>>>(edgef, W_e1, b_e1, Rbf);
  k_cvt<<<(Hh * Hh * Hh + 255) / 256, 256, 0, stream>>>(W_e2, W2b, Hh * Hh * Hh);
  k_seg<<<1, 128, 0, stream>>>(batch, seg);

  // T rounds of message passing (A recomputed in-register via WMMA each round)
  float* cur = hA; float* nxt = hB;
  for (int s = 0; s < Tsteps; ++s) {
    k_zero<<<(Nn * Hh + 255) / 256, 256, 0, stream>>>(agg, Nn * Hh);
    k_hbias<<<Nn, 64, 0, stream>>>(cur, b_e2, hbias);
    k_msg<<<Ee / 16, 256, 0, stream>>>(Rbf, W2b, hbias, cur, Esrc, Etgt, agg);
    k_gru<<<Nn / 4, 256, 0, stream>>>(agg, cur, W_ih_g, W_hh_g, b_ih_g, b_hh_g, nxt);
    float* tmp = cur; cur = nxt; nxt = tmp;
  }

  // Set2Set readout
  k_zero<<<(Bb * Hh + 255) / 256, 256, 0, stream>>>(hh, Bb * Hh);
  k_zero<<<(Bb * Hh + 255) / 256, 256, 0, stream>>>(ccv, Bb * Hh);
  k_zero<<<(Bb * 2 * Hh + 255) / 256, 256, 0, stream>>>(q_star, Bb * 2 * Hh);
  for (int s = 0; s < S2Sn; ++s) {
    k_lstm<<<Bb, 256, 0, stream>>>(q_star, hh, ccv, W_ih_l, W_hh_l, b_ih_l, b_hh_l);
    if (s + 1 < S2Sn)
      k_attn<<<Bb, 256, 0, stream>>>(cur, hh, seg, e_ws, q_star);
  }
  k_out<<<(Bb * TGTn + 255) / 256, 256, 0, stream>>>(hh, W_out, b_outp, (float*)d_out);
}
